// CrossAttention_67551245631848
// MI455X (gfx1250) — matrix operand order repaired and checked by reading, not compiled, not run on hardware
//
#include <hip/hip_runtime.h>
#include <hip/hip_bf16.h>

typedef __attribute__((ext_vector_type(16))) _Float16 v16h;
typedef __attribute__((ext_vector_type(8)))  _Float16 v8h;
typedef __attribute__((ext_vector_type(8)))  float    v8f;

#define CH 128            // all convs are 128 -> 128 channels
#define KDIM (9 * CH)     // 1152, k = tap*128 + cin
#define NSTEP 36          // 1152 / 32 K-steps
#define SCALE 0.17677669529663687f

// ---------------------------------------------------------------------------
// Repack conv weights (Cout,Cin,3,3) fp32 -> f16 GEMM A-matrix [Cout][tap*128+cin]
// ---------------------------------------------------------------------------
__global__ void prep_weights(const float* __restrict__ w, _Float16* __restrict__ A) {
    int idx = blockIdx.x * 256 + threadIdx.x;          // total 128*1152 = 147456
    if (idx >= CH * KDIM) return;
    int cout = idx / KDIM;
    int k    = idx - cout * KDIM;
    int tap  = k >> 7;                                 // 0..8
    int cin  = k & 127;
    A[idx] = (_Float16)w[(cout * CH + cin) * 9 + tap];
}

// ---------------------------------------------------------------------------
// Implicit-GEMM 3x3 conv (pad=1) via WMMA f16->f32, double-buffered LDS.
// Grid: (H*W/64, 1, B). Block: 256 = 8 waves (wave32).
// Workgroup tile: 128 cout x 64 positions. Wave w: couts [16w,16w+16) x all 64 n.
// Software pipeline: step s computes from LDS buffer p while step s+1 is
// fetched from global into registers and committed to buffer p^1; a single
// __syncthreads per K-step.
// ---------------------------------------------------------------------------
template <typename InT, typename OutT, bool RELU>
__global__ void conv3x3_wmma(const InT* __restrict__ in,
                             const _Float16* __restrict__ Aw,
                             const float* __restrict__ bias,
                             OutT* __restrict__ out,
                             int H, int W) {
    __shared__ __align__(16) _Float16 As [2][128][40];  // [buf][cout][k]
    __shared__ __align__(16) _Float16 BsT[2][64][40];   // [buf][n][k]

    const int b     = blockIdx.z;
    const int nbase = blockIdx.x * 64;        // position tile base
    const int HW    = H * W;

    const int tid  = threadIdx.x;
    const int wave = tid >> 5;                // 0..7 (wave32)
    const int lane = tid & 31;
    const int am   = wave * 16;               // wave's cout stripe
    const int m    = lane & 15;
    const int half = lane >> 4;

    // staging coordinates (loop-invariant)
    const int arow = tid >> 1;                // A: 0..127
    const int akk  = (tid & 1) * 16;          // A: k offset 0 or 16
    const int bn   = tid & 63;                // B: position within tile
    const int bkk  = (tid >> 6) * 8;          // B: k offset 0,8,16,24
    const int nglb = nbase + bn;
    const int h0   = nglb / W;
    const int w0   = nglb - h0 * W;
    const InT* const inb = in + (size_t)(b * CH + bkk) * HW;
    const _Float16* const awb = Aw + (size_t)arow * KDIM + akk;

    // fetch step s (tap = s>>2, cin block = (s&3)*32) from global into regs
    auto fetch = [&](int s, v8h& a0, v8h& a1, v8h& pb) {
        const int tap = s >> 2;
        const int c4  = s & 3;
        const v8h* ap = (const v8h*)(awb + tap * 128 + c4 * 32);
        a0 = ap[0];
        a1 = ap[1];
        const int dy = tap / 3 - 1;
        const int dx = tap - (tap / 3) * 3 - 1;
        const int sh = h0 + dy;
        const int sw = w0 + dx;
        const bool ok = (sh >= 0) & (sh < H) & (sw >= 0) & (sw < W);
        const InT* src = inb + (size_t)(c4 * 32) * HW + sh * W + sw;
#pragma unroll
        for (int j = 0; j < 8; ++j)
            pb[j] = ok ? (_Float16)(float)src[(size_t)j * HW] : (_Float16)0.0f;
        __builtin_prefetch((const void*)(src + (size_t)32 * HW), 0, 1);
    };

    v8f acc[4] = {{}, {}, {}, {}};

    // prologue: stage step 0 into buffer 0
    {
        v8h a0, a1, pb;
        fetch(0, a0, a1, pb);
        *(v8h*)&As[0][arow][akk]     = a0;
        *(v8h*)&As[0][arow][akk + 8] = a1;
        *(v8h*)&BsT[0][bn][bkk]      = pb;
    }
    __syncthreads();

    int p = 0;
    for (int s = 0; s < NSTEP; ++s) {
        // ---- fetch next step from global into registers (overlaps WMMAs) ----
        v8h na0, na1, npb;
        const bool have_next = (s + 1 < NSTEP);
        if (have_next) fetch(s + 1, na0, na1, npb);

        // ---- A fragment: two contiguous b128 chunks of one LDS row ----
        v8h fa0 = *(const v8h*)&As[p][am + m][8 * half];
        v8h fa1 = *(const v8h*)&As[p][am + m][16 + 8 * half];
        v16h af = __builtin_shufflevector(fa0, fa1, 0, 1, 2, 3, 4, 5, 6, 7,
                                          8, 9, 10, 11, 12, 13, 14, 15);
        // ---- preload all 4 B fragments, then burst 4 WMMAs ----
        v16h bf[4];
#pragma unroll
        for (int f = 0; f < 4; ++f) {
            const _Float16* brow = &BsT[p][f * 16 + m][8 * half];
            v8h b0 = *(const v8h*)brow;
            v8h b1 = *(const v8h*)(brow + 16);
            bf[f] = __builtin_shufflevector(b0, b1, 0, 1, 2, 3, 4, 5, 6, 7,
                                            8, 9, 10, 11, 12, 13, 14, 15);
        }
#pragma unroll
        for (int f = 0; f < 4; ++f)
            acc[f] = __builtin_amdgcn_wmma_f32_16x16x32_f16(false, af, false, bf[f],
                                                            (short)0, acc[f],
                                                            false, false);

        // ---- commit next step to the other buffer ----
        if (have_next) {
            const int q = p ^ 1;
            *(v8h*)&As[q][arow][akk]     = na0;
            *(v8h*)&As[q][arow][akk + 8] = na1;
            *(v8h*)&BsT[q][bn][bkk]      = npb;
        }
        __syncthreads();
        p ^= 1;
    }

    // ---- store D: lane holds N = lane&15, VGPR r holds M = r + 8*half ----
#pragma unroll
    for (int r = 0; r < 8; ++r) {
        const int cout = am + r + 8 * half;
        const float bv = bias[cout];
        size_t o = (size_t)(b * CH + cout) * HW + nbase;
#pragma unroll
        for (int f = 0; f < 4; ++f) {
            float v = acc[f][r] + bv;
            if (RELU) v = fmaxf(v, 0.0f);
            out[o + f * 16 + m] = (OutT)v;
        }
    }
}

// ---------------------------------------------------------------------------
// Per-query attention over H axis.
// q:(B,E,SS) f16, kmap/vmap:(B,E,H,W) f16, u:(B,SS) int -> a:(B,E,SS) f16
// One 128-thread block per (b, query). E == H == 128.
// ---------------------------------------------------------------------------
__global__ void attn_kernel(const _Float16* __restrict__ q,
                            const _Float16* __restrict__ kmap,
                            const _Float16* __restrict__ vmap,
                            const int* __restrict__ u,
                            _Float16* __restrict__ a,
                            int H, int W, int SS) {
    __shared__ float qs[128];
    __shared__ float red[128];
    __shared__ float attn[128];

    const int b  = blockIdx.x / SS;
    const int qn = blockIdx.x - b * SS;
    const int t  = threadIdx.x;

    int uc = u[b * SS + qn];
    uc = uc < 0 ? 0 : (uc >= W ? W - 1 : uc);

    qs[t] = (float)q[(size_t)(b * CH + t) * SS + qn];
    __syncthreads();

    // dot over E for this thread's h = t
    float d = 0.0f;
#pragma unroll 8
    for (int e = 0; e < CH; ++e)
        d += qs[e] * (float)kmap[((size_t)(b * CH + e) * H + t) * W + uc];
    d *= SCALE;

    // softmax over h (128 threads)
    red[t] = d; __syncthreads();
    for (int s = 64; s > 0; s >>= 1) {
        if (t < s) red[t] = fmaxf(red[t], red[t + s]);
        __syncthreads();
    }
    float mx = red[0]; __syncthreads();
    float ex = expf(d - mx);
    red[t] = ex; __syncthreads();
    for (int s = 64; s > 0; s >>= 1) {
        if (t < s) red[t] += red[t + s];
        __syncthreads();
    }
    float sm = red[0]; __syncthreads();
    attn[t] = ex / sm;
    __syncthreads();

    // a[e] for this thread's e = t
    float acc = 0.0f;
#pragma unroll 8
    for (int h = 0; h < 128; ++h)
        acc += attn[h] * (float)vmap[((size_t)(b * CH + t) * H + h) * W + uc];
    a[(size_t)(b * CH + t) * SS + qn] = (_Float16)acc;
}

// ---------------------------------------------------------------------------
extern "C" void kernel_launch(void* const* d_in, const int* in_sizes, int n_in,
                              void* d_out, int out_size, void* d_ws, size_t ws_size,
                              hipStream_t stream) {
    const float* x      = (const float*)d_in[0];   // (4,128,32,32)
    const float* y      = (const float*)d_in[1];   // (4,128,128,512)
    const int*   u      = (const int*)  d_in[2];   // (4,32,32)
    const float* q_w1   = (const float*)d_in[3];
    const float* q_b1   = (const float*)d_in[4];
    const float* q_w2   = (const float*)d_in[5];
    const float* q_b2   = (const float*)d_in[6];
    const float* k_w1   = (const float*)d_in[7];
    const float* k_b1   = (const float*)d_in[8];
    const float* k_w2   = (const float*)d_in[9];
    const float* k_b2   = (const float*)d_in[10];
    const float* v_w1   = (const float*)d_in[11];
    const float* v_b1   = (const float*)d_in[12];
    const float* v_w2   = (const float*)d_in[13];
    const float* v_b2   = (const float*)d_in[14];
    const float* proj_w = (const float*)d_in[15];
    const float* proj_b = (const float*)d_in[16];
    float* z = (float*)d_out;                      // (4,128,32,32) fp32

    const int B = 4, Hs = 32, Ws = 32, Hy = 128, Wy = 512;
    const int SS = Hs * Ws;                        // 1024

    // ---- workspace layout (bytes) ----
    const size_t AW_BYTES = (size_t)CH * KDIM * sizeof(_Float16);   // 294912 per weight
    char* base = (char*)d_ws;
    _Float16* Aw[7];
    for (int i = 0; i < 7; ++i) Aw[i] = (_Float16*)(base + (size_t)i * AW_BYTES);
    size_t off = 7 * AW_BYTES;
    off = (off + 255) & ~(size_t)255;
    _Float16* t16 = (_Float16*)(base + off); off += (size_t)B * CH * Hy * Wy * 2; // 64 MB temp
    _Float16* k16 = (_Float16*)(base + off); off += (size_t)B * CH * Hy * Wy * 2;
    _Float16* v16 = (_Float16*)(base + off); off += (size_t)B * CH * Hy * Wy * 2;
    _Float16* q16 = (_Float16*)(base + off); off += (size_t)B * CH * SS * 2;
    _Float16* a16 = (_Float16*)(base + off); off += (size_t)B * CH * SS * 2;

    // ---- repack all 7 weights to f16 GEMM A-matrices ----
    const float* wsrc[7] = {q_w1, q_w2, k_w1, k_w2, v_w1, v_w2, proj_w};
    for (int i = 0; i < 7; ++i)
        prep_weights<<<(CH * KDIM + 255) / 256, 256, 0, stream>>>(wsrc[i], Aw[i]);

    dim3 blk(256);
    dim3 gS(Hs * Ws / 64, 1, B);       // 32x32 grids
    dim3 gY(Hy * Wy / 64, 1, B);       // 128x512 map

    // Q path: t1 = relu(conv(x,w1)); q = conv(t1,w2)
    conv3x3_wmma<float,    _Float16, true ><<<gS, blk, 0, stream>>>(x,   Aw[0], q_b1, t16, Hs, Ws);
    conv3x3_wmma<_Float16, _Float16, false><<<gS, blk, 0, stream>>>(t16, Aw[1], q_b2, q16, Hs, Ws);
    // K path
    conv3x3_wmma<float,    _Float16, true ><<<gY, blk, 0, stream>>>(y,   Aw[2], k_b1, t16, Hy, Wy);
    conv3x3_wmma<_Float16, _Float16, false><<<gY, blk, 0, stream>>>(t16, Aw[3], k_b2, k16, Hy, Wy);
    // V path
    conv3x3_wmma<float,    _Float16, true ><<<gY, blk, 0, stream>>>(y,   Aw[4], v_b1, t16, Hy, Wy);
    conv3x3_wmma<_Float16, _Float16, false><<<gY, blk, 0, stream>>>(t16, Aw[5], v_b2, v16, Hy, Wy);
    // Attention: one block per (b, query)
    attn_kernel<<<B * SS, 128, 0, stream>>>(q16, k16, v16, u, a16, Hy, Wy, SS);
    // Output projection: relu(conv(a, proj_w)) -> fp32 d_out
    conv3x3_wmma<_Float16, float, true ><<<gS, blk, 0, stream>>>(a16, Aw[6], proj_b, z, Hs, Ws);
}
